// SelfAttention_6648609374938
// MI455X (gfx1250) — matrix operand order repaired and checked by reading, not compiled, not run on hardware
//
#include <hip/hip_runtime.h>
#include <hip/hip_bf16.h>
#include <stdint.h>

// ---------------------------------------------------------------------------
// Self-attention for MI455X (gfx1250), wave32 + v_wmma_f32_16x16x32_bf16.
//
// Roofline: ~89 GFLOP vs ~200MB HBM traffic @23.3TB/s (~9us) -> near the knee;
// all four matmul stages run on bf16 WMMA (f32 accumulate), softmax stats f32.
//
// Data movement:
//   - QKV projection casts f32 -> bf16 once; V stored transposed [n,h,d,s].
//   - Attention: block of 8 waves shares one (n,h); K/V tiles staged into LDS
//     via double-buffered GLOBAL_LOAD_ASYNC_TO_LDS_B128 (ASYNCcnt), cutting
//     K/V vector-memory traffic 8x. Fragments come from LDS as ds_load_b128.
//   - P (C-layout -> A-layout) converted through per-wave LDS scratch.
// ---------------------------------------------------------------------------

typedef __attribute__((ext_vector_type(16))) __bf16 bf16x16;
typedef __attribute__((ext_vector_type(8)))  __bf16 bf16x8;
typedef __attribute__((ext_vector_type(8)))  float  f32x8;

#define WMMA_BF16(a, b, c) \
  __builtin_amdgcn_wmma_f32_16x16x32_bf16(false, (a), false, (b), (short)0, (c), false, false)

static constexpr int N  = 4;
static constexpr int S  = 2048;
static constexpr int D  = 1024;
static constexpr int H  = 16;
static constexpr int HD = 64;
static constexpr int NKB = S / 32;                  // 64 key blocks
static constexpr float SCALE = 0.03125f;            // 1/sqrt(1024)
static constexpr size_t QKV_ELEMS = (size_t)N * H * S * HD;   // 8,388,608
static constexpr size_t QKV_BYTES = QKV_ELEMS * 2;            // 16 MB each

// ---- fragment builders ----------------------------------------------------
// A-matrix (16x32, M=lane%16): element i -> K = dbase + i,
// dbase = kstep*32 + (lane>=16 ? 16 : 0). 16 contiguous elements.
__device__ __forceinline__ bf16x16 make_A_f32(const float* __restrict__ row, int dbase) {
  bf16x16 a;
#pragma unroll
  for (int i = 0; i < 8; ++i) a[i]     = (__bf16)row[dbase + i];
#pragma unroll
  for (int i = 0; i < 8; ++i) a[8 + i] = (__bf16)row[dbase + 8 + i];
  return a;
}

__device__ __forceinline__ bf16x16 load_A_bf16(const __bf16* row, int dbase) {
  bf16x8 c0 = *(const bf16x8*)(row + dbase);
  bf16x8 c1 = *(const bf16x8*)(row + dbase + 8);
  bf16x16 a;
#pragma unroll
  for (int i = 0; i < 8; ++i) { a[i] = c0[i]; a[8 + i] = c1[i]; }
  return a;
}

// B-matrix (32x16, N=lane%16): element i -> K = kbase + i,
// kbase = kstep*32 + (lane>=16 ? 16 : 0). 16 contiguous elements.
__device__ __forceinline__ bf16x16 make_B_f32(const float* __restrict__ p) {
  bf16x16 b;
#pragma unroll
  for (int i = 0; i < 16; ++i) b[i] = (__bf16)p[i];
  return b;
}

__device__ __forceinline__ bf16x16 load_B_bf16(const __bf16* p) {
  bf16x8 c0 = *(const bf16x8*)(p);
  bf16x8 c1 = *(const bf16x8*)(p + 8);
  bf16x16 b;
#pragma unroll
  for (int i = 0; i < 8; ++i) { b[i] = c0[i]; b[8 + i] = c1[i]; }
  return b;
}

// ---------------------------------------------------------------------------
// Kernel 1: fused per-head QKV projection, f32 -> bf16.
// ---------------------------------------------------------------------------
__global__ __launch_bounds__(256) void proj_qkv(
    const float* __restrict__ vin, const float* __restrict__ kin,
    const float* __restrict__ qin,
    const float* __restrict__ Wv, const float* __restrict__ Wk,
    const float* __restrict__ Wq,
    __bf16* __restrict__ Qb, __bf16* __restrict__ Kb, __bf16* __restrict__ Vt) {
  const int lane = threadIdx.x & 31;
  const int l16  = lane & 15;
  const int hi   = lane >> 4;
  const int gid  = (blockIdx.x * blockDim.x + threadIdx.x) >> 5;   // 8192 waves
  const int h      = gid & (H - 1);
  const int rtile  = gid / H;
  const int n      = rtile / (S / 16);
  const int stile  = rtile % (S / 16);
  const int s0     = stile * 16;

  const size_t row_in = ((size_t)(n * S + s0 + l16)) * D + h * HD;
  const int nh = n * H + h;

  const float* xs[3] = { qin + row_in, kin + row_in, vin + row_in };
  const float* Ws[3] = { Wq, Wk, Wv };

#pragma unroll
  for (int w = 0; w < 3; ++w) {
    bf16x16 a0 = make_A_f32(xs[w], 0 * 32 + hi * 16);
    bf16x16 a1 = make_A_f32(xs[w], 1 * 32 + hi * 16);
#pragma unroll
    for (int et = 0; et < 4; ++et) {
      const int e = et * 16 + l16;
      f32x8 acc = {};
      acc = WMMA_BF16(a0, make_B_f32(Ws[w] + e * HD + 0 * 32 + hi * 16), acc);
      acc = WMMA_BF16(a1, make_B_f32(Ws[w] + e * HD + 1 * 32 + hi * 16), acc);
      if (w < 2) {
        __bf16* dst = (w == 0 ? Qb : Kb);
#pragma unroll
        for (int i = 0; i < 8; ++i) {
          const int s = s0 + i + 8 * hi;
          dst[((size_t)nh * S + s) * HD + e] = (__bf16)acc[i];
        }
      } else {
        bf16x8 pk;
#pragma unroll
        for (int i = 0; i < 8; ++i) pk[i] = (__bf16)acc[i];
        *(bf16x8*)(Vt + ((size_t)nh * HD + e) * S + s0 + 8 * hi) = pk;
      }
    }
  }
}

// ---------------------------------------------------------------------------
// Kernel 2: flash attention. Block = 8 waves sharing one (n,h); K/V tiles
// double-buffered in LDS via async-to-LDS loads; online softmax per wave.
// ---------------------------------------------------------------------------
__global__ __launch_bounds__(256) void attn(
    const __bf16* __restrict__ Qb, const __bf16* __restrict__ Kb,
    const __bf16* __restrict__ Vt, __bf16* __restrict__ Xb) {
  __shared__ __align__(16) __bf16 Ks[2][32 * 64];     // [key][d]     4KB x2
  __shared__ __align__(16) __bf16 Vs[2][64 * 32];     // [d][key]     4KB x2
  __shared__ __align__(16) __bf16 plds[8][16 * 32];   // per-wave P   1KB x8

  const int tid  = threadIdx.x;
  const int lane = tid & 31;
  const int l16  = lane & 15;
  const int hi   = lane >> 4;
  const int wave = tid >> 5;

  const int nh     = blockIdx.x >> 4;   // 0..63  (n*H+h)
  const int qblock = blockIdx.x & 15;   // 0..15
  const int qt     = qblock * 8 + wave; // 0..127
  const int n = nh / H, h = nh % H;

  const __bf16* qbase = Qb + ((size_t)nh * S + qt * 16) * HD;
  const __bf16* kbase = Kb + (size_t)nh * S * HD;
  const __bf16* vbase = Vt + (size_t)nh * HD * S;
  __bf16* lp = plds[wave];

  // cooperative staging: each of 256 threads moves one 16B chunk of K and V
  const int krow = tid >> 3, kchk = tid & 7;   // K: 32 rows x 8 chunks of 8 bf16
  const int vrow = tid >> 2, vchk = tid & 3;   // V: 64 rows x 4 chunks of 8 bf16

  auto issue = [&](int kb, int buf) {
    uint32_t koff = (uint32_t)(uintptr_t)&Ks[buf][krow * 64 + kchk * 8];
    const __bf16* kg = kbase + (size_t)(kb * 32 + krow) * HD + kchk * 8;
    asm volatile("global_load_async_to_lds_b128 %0, %1, off"
                 :: "v"(koff), "v"((unsigned long long)(uintptr_t)kg) : "memory");
    uint32_t voff = (uint32_t)(uintptr_t)&Vs[buf][vrow * 32 + vchk * 8];
    const __bf16* vg = vbase + (size_t)vrow * S + (size_t)kb * 32 + vchk * 8;
    asm volatile("global_load_async_to_lds_b128 %0, %1, off"
                 :: "v"(voff), "v"((unsigned long long)(uintptr_t)vg) : "memory");
  };

  // Q fragments held in registers for the whole key loop
  bf16x16 aQ0 = load_A_bf16(qbase + l16 * HD, 0 * 32 + hi * 16);
  bf16x16 aQ1 = load_A_bf16(qbase + l16 * HD, 1 * 32 + hi * 16);

  float mi[8], li[8];
  f32x8 o[4] = {};
#pragma unroll
  for (int i = 0; i < 8; ++i) { mi[i] = -3.0e38f; li[i] = 0.0f; }

  issue(0, 0);
  for (int kb = 0; kb < NKB; ++kb) {
    const int cur = kb & 1;
    asm volatile("s_wait_asynccnt 0" ::: "memory");   // our async loads landed
    __syncthreads();                                  // whole tile valid in LDS
    if (kb + 1 < NKB) issue(kb + 1, cur ^ 1);         // prefetch overlaps compute

    const __bf16* ksb = &Ks[cur][0];
    const __bf16* vsb = &Vs[cur][0];

    // ---- energies: two 16x16 tiles (key cols 0-15 / 16-31) ----
    f32x8 e0 = {}, e1 = {};
    e0 = WMMA_BF16(aQ0, load_B_bf16(ksb + (0 * 16 + l16) * 64 + 0 * 32 + hi * 16), e0);
    e0 = WMMA_BF16(aQ1, load_B_bf16(ksb + (0 * 16 + l16) * 64 + 1 * 32 + hi * 16), e0);
    e1 = WMMA_BF16(aQ0, load_B_bf16(ksb + (1 * 16 + l16) * 64 + 0 * 32 + hi * 16), e1);
    e1 = WMMA_BF16(aQ1, load_B_bf16(ksb + (1 * 16 + l16) * 64 + 1 * 32 + hi * 16), e1);
#pragma unroll
    for (int i = 0; i < 8; ++i) { e0[i] *= SCALE; e1[i] *= SCALE; }

    // ---- row max over 32 keys (xor-reduce within 16-lane halves) ----
    float rm[8];
#pragma unroll
    for (int i = 0; i < 8; ++i) rm[i] = fmaxf(e0[i], e1[i]);
#pragma unroll
    for (int m = 1; m <= 8; m <<= 1)
#pragma unroll
      for (int i = 0; i < 8; ++i) rm[i] = fmaxf(rm[i], __shfl_xor(rm[i], m, 32));

    float alpha[8];
#pragma unroll
    for (int i = 0; i < 8; ++i) {
      float mn = fmaxf(mi[i], rm[i]);
      alpha[i] = __expf(mi[i] - mn);
      mi[i] = mn;
    }
#pragma unroll
    for (int i = 0; i < 8; ++i) {
      e0[i] = __expf(e0[i] - mi[i]);
      e1[i] = __expf(e1[i] - mi[i]);
    }
    float rs[8];
#pragma unroll
    for (int i = 0; i < 8; ++i) rs[i] = e0[i] + e1[i];
#pragma unroll
    for (int m = 1; m <= 8; m <<= 1)
#pragma unroll
      for (int i = 0; i < 8; ++i) rs[i] += __shfl_xor(rs[i], m, 32);
#pragma unroll
    for (int i = 0; i < 8; ++i) li[i] = li[i] * alpha[i] + rs[i];
#pragma unroll
    for (int t = 0; t < 4; ++t)
#pragma unroll
      for (int i = 0; i < 8; ++i) o[t][i] *= alpha[i];

    // ---- P: C-layout regs -> LDS (row-major 16x32) -> A-layout fragment ----
#pragma unroll
    for (int i = 0; i < 8; ++i) {
      const int m = i + 8 * hi;
      lp[m * 32 + l16]      = (__bf16)e0[i];
      lp[m * 32 + 16 + l16] = (__bf16)e1[i];
    }
    asm volatile("s_wait_dscnt 0" ::: "memory");
    bf16x16 aP = load_A_bf16(lp + l16 * 32, hi * 16);

    // ---- O += P(16x32) x V(32x64) from LDS-staged transposed V ----
#pragma unroll
    for (int t = 0; t < 4; ++t)
      o[t] = WMMA_BF16(aP, load_B_bf16(vsb + (t * 16 + l16) * 32 + hi * 16), o[t]);
  }

  // ---- epilogue: normalize, store Xb bf16 [n,s,1024] ----
  float inv[8];
#pragma unroll
  for (int i = 0; i < 8; ++i) inv[i] = 1.0f / li[i];
#pragma unroll
  for (int t = 0; t < 4; ++t)
#pragma unroll
    for (int i = 0; i < 8; ++i) {
      const int q = qt * 16 + i + 8 * hi;
      Xb[((size_t)n * S + q) * D + h * HD + t * 16 + l16] = (__bf16)(o[t][i] * inv[i]);
    }
}

// ---------------------------------------------------------------------------
// Kernel 3: output projection Y = X @ Wo^T + bo. Wave task: 16x16 tile, K=1024.
// ---------------------------------------------------------------------------
__global__ __launch_bounds__(256) void out_gemm(
    const __bf16* __restrict__ Xb, const float* __restrict__ Wo,
    const float* __restrict__ bo, float* __restrict__ Y) {
  const int lane = threadIdx.x & 31;
  const int l16  = lane & 15;
  const int hi   = lane >> 4;
  const int gid  = (blockIdx.x * blockDim.x + threadIdx.x) >> 5;   // 32768 waves
  const int et = gid & (D / 16 - 1);
  const int rt = gid / (D / 16);
  const int e  = et * 16 + l16;

  const __bf16* arow = Xb + (size_t)(rt * 16 + l16) * D;
  const float*  wrow = Wo + (size_t)e * D;

  f32x8 acc = {};
#pragma unroll 4
  for (int ks = 0; ks < D / 32; ++ks) {
    bf16x16 a = load_A_bf16(arow, ks * 32 + hi * 16);
    bf16x16 b = make_B_f32(wrow + ks * 32 + hi * 16);
    acc = WMMA_BF16(a, b, acc);
  }
  const float bias = bo[e];
#pragma unroll
  for (int i = 0; i < 8; ++i) {
    const int r = rt * 16 + i + 8 * hi;
    Y[(size_t)r * D + e] = acc[i] + bias;
  }
}

// ---------------------------------------------------------------------------
extern "C" void kernel_launch(void* const* d_in, const int* in_sizes, int n_in,
                              void* d_out, int out_size, void* d_ws, size_t ws_size,
                              hipStream_t stream) {
  const float* values = (const float*)d_in[0];
  const float* keys   = (const float*)d_in[1];
  const float* query  = (const float*)d_in[2];
  const float* Wv     = (const float*)d_in[3];
  const float* Wk     = (const float*)d_in[4];
  const float* Wq     = (const float*)d_in[5];
  const float* Wo     = (const float*)d_in[6];
  const float* bo     = (const float*)d_in[7];
  float* out = (float*)d_out;

  char* ws = (char*)d_ws;
  __bf16* Qb = (__bf16*)(ws);
  __bf16* Kb = (__bf16*)(ws + QKV_BYTES);
  __bf16* Vt = (__bf16*)(ws + 2 * QKV_BYTES);
  __bf16* Xb = (__bf16*)(ws + 3 * QKV_BYTES);

  proj_qkv<<<1024, 256, 0, stream>>>(values, keys, query, Wv, Wk, Wq, Qb, Kb, Vt);
  attn<<<1024, 256, 0, stream>>>(Qb, Kb, Vt, Xb);
  out_gemm<<<4096, 256, 0, stream>>>(Xb, Wo, bo, out);
}
